// UnimolETEmbedding_38070590112231
// MI455X (gfx1250) — compile-verified, matrix operand order checked by reading, not run on hardware
//
#include <hip/hip_runtime.h>
#include <hip/hip_bf16.h>
#include <hip/hip_fp16.h>

typedef __attribute__((ext_vector_type(16))) _Float16 v16h;
typedef __attribute__((ext_vector_type(8)))  float    v8f;

#define B_   32
#define NA_  128
#define NB_  128
#define D_   512
#define NH_  16
#define APO_ 128
#define VA_  64
#define L_   256   // NA + NB

// ---------------------------------------------------------------------------
// Kernel 1: atoms_emb [L, B, D]  (gather + add, memory bound)
// ---------------------------------------------------------------------------
__global__ __launch_bounds__(128)
void atoms_emb_kernel(const int* __restrict__ atoms, const int* __restrict__ chirals,
                      const int* __restrict__ bond_indices, const int* __restrict__ bond_values,
                      const float* __restrict__ atype_emb, const float* __restrict__ chiral_emb,
                      const float* __restrict__ btype_tab, const float* __restrict__ bond_atom_tab,
                      float* __restrict__ out0)
{
    int row = blockIdx.x;            // row = l * B + b
    int l = row / B_;
    int b = row - l * B_;
    int d = threadIdx.x * 4;

    float4 v;
    if (l < NA_) {
        int at = atoms[b * NA_ + l];
        int ch = chirals[b * NA_ + l];
        float4 a = *(const float4*)(atype_emb  + (size_t)at * D_ + d);
        float4 c = *(const float4*)(chiral_emb + (size_t)ch * D_ + d);
        v = make_float4(a.x + c.x, a.y + c.y, a.z + c.z, a.w + c.w);
    } else {
        int jb = l - NA_;
        int bv = bond_values[b * NB_ + jb];
        int e0 = bond_indices[(b * NB_ + jb) * 2 + 0];
        int e1 = bond_indices[(b * NB_ + jb) * 2 + 1];
        int a0 = atoms[b * NA_ + e0];
        int a1 = atoms[b * NA_ + e1];
        float4 t = *(const float4*)(btype_tab     + (size_t)bv * D_ + d);
        float4 u = *(const float4*)(bond_atom_tab + (size_t)a0 * D_ + d);
        float4 w = *(const float4*)(bond_atom_tab + (size_t)a1 * D_ + d);
        v = make_float4(t.x + u.x + w.x, t.y + u.y + w.y, t.z + u.z + w.z, t.w + u.w + w.w);
    }
    *(float4*)(out0 + (size_t)row * D_ + d) = v;
}

// ---------------------------------------------------------------------------
// Kernel 2: fused RBF + (524288 x 128) @ (128 x 16) via WMMA f16 -> f32.
// One wave per (b, i, 16-wide j tile). A rows = j tile (M), K = p, N = h.
// Epilogue adds lin_b + bond_emb, applies -inf column mask, writes the
// transposed [B, NH, L, L] layout directly (two contiguous float4 per lane).
// ---------------------------------------------------------------------------
__global__ __launch_bounds__(256)
void pair_core_kernel(const int* __restrict__ atoms,
                      const float* __restrict__ coords,
                      const int* __restrict__ bonds,
                      const float* __restrict__ wtab,
                      const float* __restrict__ btab,
                      const float* __restrict__ means,
                      const float* __restrict__ stds,
                      const float* __restrict__ bond_emb,
                      const float* __restrict__ linW,
                      const float* __restrict__ linb,
                      float* __restrict__ out1)
{
    __shared__ float s_mean[APO_];
    __shared__ float s_invs[APO_];
    int tid = threadIdx.x;
    if (tid < APO_) {
        float s = fabsf(stds[tid]) + 1e-5f;
        s_mean[tid] = means[tid];
        s_invs[tid] = 1.0f / s;
    }
    __syncthreads();

    int wave = (blockIdx.x * blockDim.x + tid) >> 5;   // global wave id
    int lane = tid & 31;

    // wave -> (b, i, j_base): 128 * 8 = 1024 tiles per batch element
    int b      = wave >> 10;
    int rem    = wave & 1023;
    int i      = rem >> 3;
    int j_base = (rem & 7) << 4;

    int n   = lane & 15;            // h column (B/C/D) and row id (A)
    int sel = (lane >> 4) << 3;     // 0 for lanes 0-15, 8 for lanes 16-31

    // --- B operand: lin_W (f32) -> f16, 4 K-chunks, ISA 16-bit B layout ---
    v16h bw0, bw1, bw2, bw3;
    #pragma unroll
    for (int hh = 0; hh < 16; ++hh) {
        int p = sel + (hh & 7) + ((hh >> 3) << 4);
        bw0[hh] = (_Float16)linW[(p      ) * NH_ + n];
        bw1[hh] = (_Float16)linW[(p + 32 ) * NH_ + n];
        bw2[hh] = (_Float16)linW[(p + 64 ) * NH_ + n];
        bw3[hh] = (_Float16)linW[(p + 96 ) * NH_ + n];
    }

    // --- per-lane row data: j = j_base + (lane & 15) ---
    int jl = j_base + n;
    const float* ci = coords + ((size_t)b * NA_ + i ) * 3;
    const float* cj = coords + ((size_t)b * NA_ + jl) * 3;
    float dx = cj[0] - ci[0], dy = cj[1] - ci[1], dz = cj[2] - ci[2];
    float dist = sqrtf(dx * dx + dy * dy + dz * dz);

    int ai = atoms[b * NA_ + i];
    int aj = atoms[b * NA_ + jl];
    const float* wrow = wtab + (size_t)(aj * VA_ + ai) * APO_;
    const float* brow = btab + (size_t)(aj * VA_ + ai) * APO_;

    v8f acc = {};
    #pragma unroll
    for (int k = 0; k < 4; ++k) {
        int base0 = k * 32 + sel;
        float4 w0 = *(const float4*)(wrow + base0);
        float4 w1 = *(const float4*)(wrow + base0 + 4);
        float4 w2 = *(const float4*)(wrow + base0 + 16);
        float4 w3 = *(const float4*)(wrow + base0 + 20);
        float4 c0 = *(const float4*)(brow + base0);
        float4 c1 = *(const float4*)(brow + base0 + 4);
        float4 c2 = *(const float4*)(brow + base0 + 16);
        float4 c3 = *(const float4*)(brow + base0 + 20);
        float wv[16] = {w0.x,w0.y,w0.z,w0.w, w1.x,w1.y,w1.z,w1.w,
                        w2.x,w2.y,w2.z,w2.w, w3.x,w3.y,w3.z,w3.w};
        float bv[16] = {c0.x,c0.y,c0.z,c0.w, c1.x,c1.y,c1.z,c1.w,
                        c2.x,c2.y,c2.z,c2.w, c3.x,c3.y,c3.z,c3.w};
        v16h a;
        #pragma unroll
        for (int hh = 0; hh < 16; ++hh) {
            int p = base0 + (hh & 7) + ((hh >> 3) << 4);
            float inv_s = s_invs[p];
            float t = (wv[hh] * dist + bv[hh] - s_mean[p]) * inv_s;
            float g = __expf(-0.5f * t * t) * 0.3989422804f * inv_s;
            a[hh] = (_Float16)g;
        }
        v16h bm = (k == 0) ? bw0 : (k == 1) ? bw1 : (k == 2) ? bw2 : bw3;
        acc = __builtin_amdgcn_wmma_f32_16x16x32_f16(
                  false, a, false, bm, (short)0, acc, false, false);
    }

    // --- epilogue: + lin_b + bond_emb, column padding mask, transposed store ---
    int h  = n;                     // output head for this lane
    int j0 = j_base + sel;          // this lane covers j0 .. j0+7 (C layout)
    float lb = linb[h];
    const int* bondrow = bonds + ((size_t)b * NA_ + i) * NA_;
    const int* atomrow = atoms + (size_t)b * NA_;
    const float ninf = -__builtin_huge_valf();

    float res[8];
    #pragma unroll
    for (int r = 0; r < 8; ++r) {
        int j = j0 + r;
        float v = acc[r] + lb + bond_emb[bondrow[j] * NH_ + h];
        if (atomrow[j] == 0) v = ninf;
        res[r] = v;
    }
    float* orow = out1 + (((size_t)b * NH_ + h) * L_ + i) * L_ + j0;
    *(float4*)(orow    ) = make_float4(res[0], res[1], res[2], res[3]);
    *(float4*)(orow + 4) = make_float4(res[4], res[5], res[6], res[7]);
}

// ---------------------------------------------------------------------------
// Kernel 3: apairs edge blocks (+ -inf mask), bdist_out, padding_mask
// ---------------------------------------------------------------------------
__global__ __launch_bounds__(256)
void edges_kernel(const int* __restrict__ atoms,
                  const int* __restrict__ bond_indices,
                  const int* __restrict__ bond_values,
                  const int* __restrict__ bdist,
                  const float* __restrict__ ab_tab,
                  float* __restrict__ out1,   // apairs [B,NH,L,L]
                  float* __restrict__ out2,   // bdist_out [B,L,L]
                  float* __restrict__ out3)   // padding_mask [B,L]
{
    int idx = blockIdx.x * blockDim.x + threadIdx.x;   // over B*L*L
    int b = idx >> 16;
    int r = idx & 65535;
    int i = r >> 8;
    int j = r & 255;

    bool in_core = false, use_ab = false;
    float bd = 0.0f;
    if (i < NA_ && j < NA_) {
        bd = (float)bdist[((size_t)b * NA_ + i) * NA_ + j];
        in_core = true;
    } else if (i >= NA_ && j < NA_) {
        int jb = i - NA_;
        int bv = bond_values[b * NB_ + jb];
        int e0 = bond_indices[(b * NB_ + jb) * 2 + 0];
        int e1 = bond_indices[(b * NB_ + jb) * 2 + 1];
        use_ab = (bv != 0) && (j == e0 || j == e1);
        bd = use_ab ? 8.0f : 0.0f;
    } else if (i < NA_) {          // j >= NA_
        int jb = j - NA_;
        int bv = bond_values[b * NB_ + jb];
        int e0 = bond_indices[(b * NB_ + jb) * 2 + 0];
        int e1 = bond_indices[(b * NB_ + jb) * 2 + 1];
        use_ab = (bv != 0) && (i == e0 || i == e1);
        bd = use_ab ? 8.0f : 0.0f;
    }
    out2[idx] = bd;

    bool maskj = (j < NA_) ? (atoms[b * NA_ + j] == 0)
                           : (bond_values[b * NB_ + (j - NA_)] == 0);
    const float ninf = -__builtin_huge_valf();

    if (!in_core) {
        #pragma unroll
        for (int h = 0; h < NH_; ++h) {
            float v = use_ab ? ab_tab[NH_ + h] : 0.0f;   // row 0 of ab table is zero
            if (maskj) v = ninf;
            out1[(((size_t)b * NH_ + h) * L_ + i) * L_ + j] = v;
        }
    }
    if (i == 0) out3[b * L_ + j] = maskj ? 1.0f : 0.0f;
}

// ---------------------------------------------------------------------------
extern "C" void kernel_launch(void* const* d_in, const int* in_sizes, int n_in,
                              void* d_out, int out_size, void* d_ws, size_t ws_size,
                              hipStream_t stream)
{
    const int*   atoms        = (const int*)  d_in[0];
    const int*   chirals      = (const int*)  d_in[1];
    const float* coords       = (const float*)d_in[2];
    const int*   bonds        = (const int*)  d_in[3];
    const int*   bond_indices = (const int*)  d_in[4];
    const int*   bond_values  = (const int*)  d_in[5];
    const int*   bdist        = (const int*)  d_in[6];
    const float* atype_emb    = (const float*)d_in[7];
    const float* chiral_emb   = (const float*)d_in[8];
    const float* wtab         = (const float*)d_in[9];
    const float* btab         = (const float*)d_in[10];
    const float* means        = (const float*)d_in[11];
    const float* stds         = (const float*)d_in[12];
    const float* bond_emb     = (const float*)d_in[13];
    const float* linW         = (const float*)d_in[14];
    const float* linb         = (const float*)d_in[15];
    const float* btype_tab    = (const float*)d_in[16];
    const float* bond_atom    = (const float*)d_in[17];
    const float* ab_tab       = (const float*)d_in[18];

    float* out0 = (float*)d_out;                          // atoms_emb [L,B,D]
    float* out1 = out0 + (size_t)L_ * B_ * D_;            // apairs    [B,NH,L,L]
    float* out2 = out1 + (size_t)B_ * NH_ * L_ * L_;      // bdist_out [B,L,L]
    float* out3 = out2 + (size_t)B_ * L_ * L_;            // padding_mask [B,L]

    atoms_emb_kernel<<<L_ * B_, 128, 0, stream>>>(
        atoms, chirals, bond_indices, bond_values,
        atype_emb, chiral_emb, btype_tab, bond_atom, out0);

    // 32 * 128 * 8 = 32768 waves, 8 waves per 256-thread block -> 4096 blocks
    pair_core_kernel<<<4096, 256, 0, stream>>>(
        atoms, coords, bonds, wtab, btab, means, stds,
        bond_emb, linW, linb, out1);

    // B*L*L / 256 = 8192 blocks
    edges_kernel<<<8192, 256, 0, stream>>>(
        atoms, bond_indices, bond_values, bdist, ab_tab, out1, out2, out3);
}
